// TTTLayer_56556129354411
// MI455X (gfx1250) — matrix operand order repaired and checked by reading, not compiled, not run on hardware
//
#include <hip/hip_runtime.h>

#define H_DIM   768
#define NH      12
#define HD      64
#define MB      16
#define KCONV   4
#define BATCH   4
#define NSEQ    2048
#define NMB     (NSEQ / MB)      // 128
#define NTOK    (BATCH * NSEQ)   // 8192
#define EPSF    1e-5f

typedef __bf16 bf16;
typedef __attribute__((ext_vector_type(16))) __bf16 v16bf;
typedef __attribute__((ext_vector_type(8)))  __bf16 v8bf;
typedef __attribute__((ext_vector_type(8)))  float  v8f;

union FragU { v16bf v; v8bf h2[2]; };

__device__ __forceinline__ v8f wmma_bf16(v16bf a, v16bf b, v8f c) {
    // v_wmma_f32_16x16x32_bf16  D = A x B + C
    return __builtin_amdgcn_wmma_f32_16x16x32_bf16(
        false, a, false, b, (short)0, c, false, false);
}

// CDNA5 async global->LDS copy (ASYNCcnt-tracked). dst: per-lane LDS byte
// address (low 32 bits of generic pointer), src: per-lane global address.
__device__ __forceinline__ void async_copy_b128(void* lds_dst, const void* gsrc) {
    unsigned lds_off = (unsigned)(unsigned long long)(size_t)lds_dst;
    asm volatile("global_load_async_to_lds_b128 %0, %1, off"
                 :: "v"(lds_off), "v"(gsrc) : "memory");
}
template <int N>
__device__ __forceinline__ void wait_asynccnt() {
    asm volatile("s_wait_asynccnt %0" :: "n"(N) : "memory");
}

// ---------------------------------------------------------------------------
// One-time prep: fp32 -> bf16 (8 elems/thread, b128 stores)
// ---------------------------------------------------------------------------
__global__ __launch_bounds__(256)
void cvt_bf16_kernel(const float* __restrict__ in, bf16* __restrict__ out)
{
    int i = blockIdx.x * 256 + threadIdx.x;
    const float4* p = (const float4*)in + (size_t)i * 2;
    float4 a = p[0], b = p[1];
    v8bf o;
    o[0] = (__bf16)a.x; o[1] = (__bf16)a.y; o[2] = (__bf16)a.z; o[3] = (__bf16)a.w;
    o[4] = (__bf16)b.x; o[5] = (__bf16)b.y; o[6] = (__bf16)b.z; o[7] = (__bf16)b.w;
    *(v8bf*)&out[(size_t)i * 8] = o;
}

// One-time prep: weight [k][n] fp32 -> transposed bf16 [n][k]
__global__ __launch_bounds__(256)
void wT_kernel(const float* __restrict__ W, bf16* __restrict__ WT)
{
    int i = blockIdx.x * 256 + threadIdx.x;       // 589824 threads
    int n = i / H_DIM, k = i % H_DIM;
    WT[i] = (__bf16)W[(size_t)k * H_DIM + n];
}

// ---------------------------------------------------------------------------
// bf16 GEMM, fixed K=N=768:  C[M,768] = A[M,768] @ BT[768,768]^T
// A row-major bf16; BT is B transposed ([n][k] bf16) so both tiles are
// byte-copied with GLOBAL_LOAD_ASYNC_TO_LDS_B128, double-buffered in LDS.
// 128x128 tile, BK=32, 256 threads = 8 waves (2x4), 8 wmma / wave / K-step.
// ---------------------------------------------------------------------------
#define GK 768
__global__ __launch_bounds__(256)
void gemm768_kernel(const bf16* __restrict__ A, const bf16* __restrict__ BT,
                    float* __restrict__ C)
{
    __shared__ __bf16 As[2][128 * 40];   // [row][k], pad 40
    __shared__ __bf16 Bs[2][128 * 40];   // [col][k], pad 40

    const int tx   = threadIdx.x;
    const int wave = tx >> 5, lane = tx & 31;
    const int wm   = wave >> 2, wn = wave & 3;
    const int lm   = lane & 15;
    const bool hi  = lane >= 16;
    const int lo   = hi ? 8 : 0;
    const int ro   = hi ? 8 : 0;
    const int bo   = hi ? 16 : 0;
    const long bm  = (long)blockIdx.x * 128, bn = (long)blockIdx.y * 128;

    // per-thread staging: 2 16B chunks per tile (512 chunks = 128 rows x 4 segs)
    auto issue = [&](int buf, int k0) {
        for (int c = 0; c < 2; ++c) {
            int id  = tx * 2 + c;
            int row = id >> 2, seg = (id & 3) * 8;
            async_copy_b128(&As[buf][row * 40 + seg], &A[(bm + row) * GK + k0 + seg]);
            async_copy_b128(&Bs[buf][row * 40 + seg], &BT[(bn + row) * GK + k0 + seg]);
        }
    };

    v8f acc[4][2];
    for (int mi = 0; mi < 4; ++mi)
        for (int ni = 0; ni < 2; ++ni)
            for (int r = 0; r < 8; ++r) acc[mi][ni][r] = 0.0f;

    issue(0, 0);
    for (int kk = 0; kk < GK / 32; ++kk) {
        int buf = kk & 1;
        if (kk + 1 < GK / 32) {
            issue(buf ^ 1, (kk + 1) * 32);
            wait_asynccnt<4>();              // newest 4 may stay in flight
        } else {
            wait_asynccnt<0>();
        }
        __syncthreads();

        v16bf af[4], bfg[2];
        for (int mi = 0; mi < 4; ++mi) {
            int row = wm * 64 + mi * 16 + lm;
            FragU u;
            u.h2[0] = *(const v8bf*)&As[buf][row * 40 + lo];
            u.h2[1] = *(const v8bf*)&As[buf][row * 40 + 16 + lo];
            af[mi] = u.v;
        }
        for (int ni = 0; ni < 2; ++ni) {
            int col = wn * 32 + ni * 16 + lm;
            FragU u;
            u.h2[0] = *(const v8bf*)&Bs[buf][col * 40 + lo];
            u.h2[1] = *(const v8bf*)&Bs[buf][col * 40 + 16 + lo];
            bfg[ni] = u.v;
        }
        for (int mi = 0; mi < 4; ++mi)
            for (int ni = 0; ni < 2; ++ni)
                acc[mi][ni] = wmma_bf16(af[mi], bfg[ni], acc[mi][ni]);
        __syncthreads();                      // protect buf before next overwrite
    }

    float* cbase = C + (size_t)(bm + wm * 64 + ro) * GK + bn + wn * 32 + lm;
    for (int mi = 0; mi < 4; ++mi)
        for (int ni = 0; ni < 2; ++ni) {
            float* cp = cbase + mi * 16 * GK + ni * 16;
            for (int r = 0; r < 8; ++r) cp[r * GK] = acc[mi][ni][r];
        }
}

// ---------------------------------------------------------------------------
// Causal depthwise conv (K=4, strictly past) + bias + RoPE + relayout to
// scan layout [B][NH][NMB][MB][HD]; also relayouts XV.
// ---------------------------------------------------------------------------
__global__ void convrope_kernel(const float* __restrict__ xqk,
                                const float* __restrict__ xvg,
                                const int* __restrict__ pos_ids,
                                const float* __restrict__ cqk, const float* __restrict__ cqb,
                                const float* __restrict__ ckk, const float* __restrict__ ckb,
                                float* __restrict__ XQs, float* __restrict__ XKs,
                                float* __restrict__ XVs)
{
    int token = blockIdx.x;
    int idx   = blockIdx.y * 128 + threadIdx.x;   // 0..383
    int h = idx >> 5, j = idx & 31;
    int b = token >> 11, n = token & 2047;
    int c0 = h * 64 + 2 * j;

    float q0 = cqb[c0], q1 = cqb[c0 + 1];
    float k0 = ckb[c0], k1 = ckb[c0 + 1];
    for (int k = 0; k < KCONV; ++k) {
        int nn = n - KCONV + k;
        if (nn >= 0) {
            long tt = (long)(b * NSEQ + nn) * H_DIM;
            float x0 = xqk[tt + c0], x1 = xqk[tt + c0 + 1];
            q0 += x0 * cqk[k * H_DIM + c0];
            q1 += x1 * cqk[k * H_DIM + c0 + 1];
            k0 += x0 * ckk[k * H_DIM + c0];
            k1 += x1 * ckk[k * H_DIM + c0 + 1];
        }
    }
    int p = pos_ids[token] & (MB - 1);
    float fj = __powf(10000.0f, -(2.0f * j) / 64.0f);
    float sn, cs;
    __sincosf((float)p * fj, &sn, &cs);
    float xq0 = q0 * cs - q1 * sn, xq1 = q0 * sn + q1 * cs;
    float xk0 = k0 * cs - k1 * sn, xk1 = k0 * sn + k1 * cs;

    int nm = n >> 4, m = n & 15;
    long off = (((long)(b * NH + h) * NMB + nm) * MB + m) * HD + 2 * j;
    XQs[off] = xq0; XQs[off + 1] = xq1;
    XKs[off] = xk0; XKs[off + 1] = xk1;
    XVs[off] = xvg[(long)token * H_DIM + c0];
    XVs[off + 1] = xvg[(long)token * H_DIM + c0 + 1];
}

// ---------------------------------------------------------------------------
// lr[b][h][nm][m] = sigmoid(dot(hs[token], lr_w[h]) + lr_b[h])
// ---------------------------------------------------------------------------
__global__ void lr_kernel(const float* __restrict__ hs, const float* __restrict__ lrw,
                          const float* __restrict__ lrbias, float* __restrict__ lr_out)
{
    int i = blockIdx.x * 256 + threadIdx.x;       // exactly 98304 threads
    int m  = i & 15;
    int nm = (i >> 4) & (NMB - 1);
    int bh = i >> 11;
    int h = bh % NH, b = bh / NH;
    long token = (long)b * NSEQ + nm * MB + m;
    const float* x = hs + token * H_DIM;
    const float* w = lrw + h * H_DIM;
    float acc = lrbias[h];
    for (int k = 0; k < H_DIM; ++k) acc += x[k] * w[k];
    lr_out[i] = 1.0f / (1.0f + __expf(-acc));
}

// ---------------------------------------------------------------------------
// TTT scan: one wave per (b,h); 128 sequential mini-batch steps.
// xq/xk/xv tiles double-buffered in LDS and prefetched for step nm+1 with
// async b128 copies while step nm's 42 WMMAs run. W1 state = 16 f32 C/D
// fragments with a bf16 LDS shadow supplying lane-local K=32 B-fragments.
// ---------------------------------------------------------------------------
__global__ __launch_bounds__(32)
void ttt_scan_kernel(const float* __restrict__ XQs, const float* __restrict__ XKs,
                     const float* __restrict__ XVs, const float* __restrict__ lr_all,
                     const float* __restrict__ lti,
                     const float* __restrict__ gamma, const float* __restrict__ beta,
                     const float* __restrict__ W1_0, const float* __restrict__ b1_0,
                     float* __restrict__ Y)
{
    const int bh = blockIdx.x;
    const int b = bh / NH, h = bh % NH;
    const int lane = threadIdx.x;
    const int lm = lane & 15;
    const bool hi = lane >= 16;
    const int lo = hi ? 8 : 0;       // A-frag K offset / row-half offset
    const int bo = hi ? 16 : 0;      // B-frag K offset

    __shared__ float sq[2][16 * 68], sk[2][16 * 68], sv[2][16 * 68];
    __shared__ float sattn[16 * 20];
    __shared__ __bf16 w1t[64 * 72];  // W1 transposed [e][d] bf16
    __shared__ __bf16 gdt[64 * 24];  // grad transposed [e][m] bf16

    float gv[4], bv[4];
    for (int j = 0; j < 4; ++j) {
        int col = j * 16 + lm;
        gv[j] = gamma[h * HD + col];
        bv[j] = beta[h * HD + col];
    }
    float tokr  = fmaxf(lti[lm] + 1.0f / (float)(lm + 1), 0.0f);
    float tok15 = fmaxf(lti[15] + 1.0f / 16.0f, 0.0f);

    v8f W1f[4][4];
    for (int i = 0; i < 4; ++i)
        for (int j = 0; j < 4; ++j) {
            int col = j * 16 + lm;
            for (int r = 0; r < 8; ++r)
                W1f[i][j][r] = W1_0[(h * HD + i * 16 + r + lo) * HD + col];
        }
    float b1v[4];
    for (int j = 0; j < 4; ++j) b1v[j] = b1_0[h * HD + j * 16 + lm];

    auto storeW1T = [&]() {
        for (int i = 0; i < 4; ++i)
            for (int j = 0; j < 4; ++j) {
                int e = j * 16 + lm;
                for (int r = 0; r < 8; ++r)
                    w1t[e * 72 + i * 16 + r + lo] = (__bf16)W1f[i][j][r];
            }
    };
    storeW1T();

    const long base = (long)(b * NH + h) * NMB * MB * HD;
    const float* lrb = lr_all + (long)(b * NH + h) * NMB * MB;

    // async prefetch of one step's three 16x64 f32 tiles (24 b128 per lane-set)
    auto issue_tiles = [&](int buf, int nm2) {
        const float* gq = XQs + base + (long)nm2 * MB * HD;
        const float* gk = XKs + base + (long)nm2 * MB * HD;
        const float* gx = XVs + base + (long)nm2 * MB * HD;
        for (int i = 0; i < 8; ++i) {
            int f = lane + 32 * i;          // 256 float4 chunks
            int r = f >> 4, c = (f & 15) * 4;
            async_copy_b128(&sq[buf][r * 68 + c], gq + f * 4);
            async_copy_b128(&sk[buf][r * 68 + c], gk + f * 4);
            async_copy_b128(&sv[buf][r * 68 + c], gx + f * 4);
        }
    };
    issue_tiles(0, 0);

    for (int nm = 0; nm < NMB; ++nm) {
        int cur = nm & 1;
        if (nm + 1 < NMB) {
            issue_tiles(cur ^ 1, nm + 1);
            wait_asynccnt<24>();             // current buffer complete
        } else {
            wait_asynccnt<0>();
        }
        const float* sqc = sq[cur];
        const float* skc = sk[cur];
        const float* svc = sv[cur];

        float lrv8[8];
        for (int e = 0; e < 8; ++e) lrv8[e] = lrb[nm * MB + lo + e];

        // A-fragments of xq/xk (K=64 in 2 chunks of 32) and B-fragments of xk^T
        v16bf qa[2], ka[2], kb[2];
        for (int ki = 0; ki < 2; ++ki) {
            for (int e = 0; e < 8; ++e) {
                qa[ki][e]     = (__bf16)sqc[lm * 68 + ki * 32 + bo + e];
                qa[ki][e + 8] = (__bf16)sqc[lm * 68 + ki * 32 + bo + 8 + e];
                ka[ki][e]     = (__bf16)skc[lm * 68 + ki * 32 + bo + e];
                ka[ki][e + 8] = (__bf16)skc[lm * 68 + ki * 32 + bo + 8 + e];
            }
            for (int e = 0; e < 16; ++e)
                kb[ki][e] = (__bf16)skc[lm * 68 + ki * 32 + bo + e];
        }

        // Z1 = xk @ W1c + b1c
        v8f z1[4];
        for (int j = 0; j < 4; ++j)
            for (int r = 0; r < 8; ++r) z1[j][r] = b1v[j];
        for (int ki = 0; ki < 2; ++ki)
            for (int j = 0; j < 4; ++j) {
                int e0 = j * 16 + lm;
                FragU u;
                u.h2[0] = *(const v8bf*)&w1t[e0 * 72 + ki * 32 + bo];
                u.h2[1] = *(const v8bf*)&w1t[e0 * 72 + ki * 32 + bo + 8];
                z1[j] = wmma_bf16(ka[ki], u.v, z1[j]);
            }

        // grad = ln_l2_bwd(Z1, xv - xk)
        float s1 = 0.f, s2 = 0.f;
        for (int j = 0; j < 4; ++j)
            for (int r = 0; r < 8; ++r) { float z = z1[j][r]; s1 += z; s2 += z * z; }
        for (int msk = 1; msk < 16; msk <<= 1) {
            s1 += __shfl_xor(s1, msk, 32);
            s2 += __shfl_xor(s2, msk, 32);
        }
        float mu = s1 / 64.0f;
        float stdv = sqrtf(s2 / 64.0f - mu * mu + EPSF);
        v8f gy[4];
        float g1 = 0.f, g2 = 0.f;
        for (int j = 0; j < 4; ++j)
            for (int r = 0; r < 8; ++r) {
                int row = r + lo, col = j * 16 + lm;
                float tgt = svc[row * 68 + col] - skc[row * 68 + col];
                float xhv = (z1[j][r] - mu) / stdv;
                float gyv = (gv[j] * xhv + bv[j] - tgt) * gv[j];
                gy[j][r] = gyv;
                g1 += gyv; g2 += gyv * xhv;
            }
        for (int msk = 1; msk < 16; msk <<= 1) {
            g1 += __shfl_xor(g1, msk, 32);
            g2 += __shfl_xor(g2, msk, 32);
        }
        v8f grad[4];
        for (int j = 0; j < 4; ++j)
            for (int r = 0; r < 8; ++r) {
                float xhv = (z1[j][r] - mu) / stdv;
                grad[j][r] = (64.0f * gy[j][r] - g1 - xhv * g2) / (64.0f * stdv);
            }

        // grad -> LDS transposed bf16, then K=16(pad 32) B-fragments
        for (int j = 0; j < 4; ++j) {
            int e = j * 16 + lm;
            for (int r = 0; r < 8; ++r) gdt[e * 24 + r + lo] = (__bf16)grad[j][r];
        }
        v16bf gb[4];
        for (int j = 0; j < 4; ++j) {
            int e0 = j * 16 + lm;
            FragU u;
            u.h2[0] = *(const v8bf*)&gdt[e0 * 24 + lo];
            u.h2[1] = *(const v8bf*)&gdt[e0 * 24 + 8];
            for (int e = 8; e < 16; ++e) u.v[e] = (__bf16)0.0f;
            gb[j] = u.v;
        }

        // attn = xq @ xk^T (one 16x16 tile), bounce via LDS for transpose
        v8f att;
        for (int r = 0; r < 8; ++r) att[r] = 0.0f;
        att = wmma_bf16(qa[0], kb[0], att);
        att = wmma_bf16(qa[1], kb[1], att);
        for (int r = 0; r < 8; ++r) sattn[(r + lo) * 20 + lm] = att[r];

        // negated, tril-masked eta A-fragments (K=16 pad 32): row i = lm
        v16bf netA, neaA;
        float etr = tokr / (float)HD;
        for (int e = 0; e < 8; ++e) {
            int jj = lo + e;
            float etv = (jj <= lm) ? etr * lrv8[e] : 0.0f;
            netA[e]     = (__bf16)(-etv);
            neaA[e]     = (__bf16)(-etv * sattn[lm * 20 + jj]);
            netA[e + 8] = (__bf16)0.0f;
            neaA[e + 8] = (__bf16)0.0f;
        }

        // Z1_bar = xq@W1c - (et*attn)@grad + b1c - (tril*et)@grad
        v8f zb[4];
        for (int j = 0; j < 4; ++j)
            for (int r = 0; r < 8; ++r) zb[j][r] = b1v[j];
        for (int j = 0; j < 4; ++j) zb[j] = wmma_bf16(netA, gb[j], zb[j]);
        for (int ki = 0; ki < 2; ++ki)
            for (int j = 0; j < 4; ++j) {
                int e0 = j * 16 + lm;
                FragU u;
                u.h2[0] = *(const v8bf*)&w1t[e0 * 72 + ki * 32 + bo];
                u.h2[1] = *(const v8bf*)&w1t[e0 * 72 + ki * 32 + bo + 8];
                zb[j] = wmma_bf16(qa[ki], u.v, zb[j]);
            }
        for (int j = 0; j < 4; ++j) zb[j] = wmma_bf16(neaA, gb[j], zb[j]);

        // y = xq + LN(Z1_bar)
        s1 = 0.f; s2 = 0.f;
        for (int j = 0; j < 4; ++j)
            for (int r = 0; r < 8; ++r) { float z = zb[j][r]; s1 += z; s2 += z * z; }
        for (int msk = 1; msk < 16; msk <<= 1) {
            s1 += __shfl_xor(s1, msk, 32);
            s2 += __shfl_xor(s2, msk, 32);
        }
        float mu2 = s1 / 64.0f;
        float std2 = sqrtf(s2 / 64.0f - mu2 * mu2 + EPSF);
        float* yp = Y + (size_t)(b * NSEQ + nm * MB) * H_DIM + h * HD;
        for (int j = 0; j < 4; ++j) {
            int col = j * 16 + lm;
            for (int r = 0; r < 8; ++r) {
                float yv = sqc[(r + lo) * 68 + col]
                         + ((zb[j][r] - mu2) / std2) * gv[j] + bv[j];
                yp[(r + lo) * H_DIM + col] = yv;
            }
        }

        // b1 update
        float l15 = tok15 / (float)HD;
        for (int j = 0; j < 4; ++j) {
            float p = 0.f;
            for (int r = 0; r < 8; ++r) p += l15 * lrv8[r] * grad[j][r];
            p += __shfl_xor(p, 16, 32);
            b1v[j] -= p;
        }

        // W1 update: W1 -= (last*xk)^T @ grad
        for (int i = 0; i < 4; ++i) {
            v16bf wu;
            int d = i * 16 + lm;
            for (int e = 0; e < 8; ++e) {
                wu[e]     = (__bf16)(-(l15 * lrv8[e]) * skc[(lo + e) * 68 + d]);
                wu[e + 8] = (__bf16)0.0f;
            }
            for (int j = 0; j < 4; ++j)
                W1f[i][j] = wmma_bf16(wu, gb[j], W1f[i][j]);
        }
        storeW1T();
    }
}

// ---------------------------------------------------------------------------
// post-norm over 768 + tanh-GELU gate:  tmp = (bf16)(gelu(G) * z)
// ---------------------------------------------------------------------------
__global__ __launch_bounds__(256)
void postgate_kernel(const float* __restrict__ Yin, const float* __restrict__ Gin,
                     const float* __restrict__ ps, const float* __restrict__ pb,
                     bf16* __restrict__ tmp)
{
    __shared__ float red[256];
    int t = blockIdx.x, tx = threadIdx.x;
    float v[3];
    float s = 0.f;
    for (int i = 0; i < 3; ++i) { v[i] = Yin[(long)t * H_DIM + tx + i * 256]; s += v[i]; }
    red[tx] = s; __syncthreads();
    for (int off = 128; off > 0; off >>= 1) {
        if (tx < off) red[tx] += red[tx + off];
        __syncthreads();
    }
    float mu = red[0] / (float)H_DIM;
    __syncthreads();
    s = 0.f;
    for (int i = 0; i < 3; ++i) { float d = v[i] - mu; s += d * d; }
    red[tx] = s; __syncthreads();
    for (int off = 128; off > 0; off >>= 1) {
        if (tx < off) red[tx] += red[tx + off];
        __syncthreads();
    }
    float stdv = sqrtf(red[0] / (float)H_DIM + EPSF);
    for (int i = 0; i < 3; ++i) {
        int c = tx + i * 256;
        float z = (v[i] - mu) / stdv * ps[c] + pb[c];
        float x = Gin[(long)t * H_DIM + c];
        float g = 0.5f * x * (1.0f + tanhf(0.7978845608028654f * (x + 0.044715f * x * x * x)));
        tmp[(long)t * H_DIM + c] = (__bf16)(g * z);
    }
}

// ---------------------------------------------------------------------------
extern "C" void kernel_launch(void* const* d_in, const int* in_sizes, int n_in,
                              void* d_out, int out_size, void* d_ws, size_t ws_size,
                              hipStream_t stream)
{
    const float* hs  = (const float*)d_in[0];
    const int*   pos = (const int*)  d_in[1];
    const float* wq  = (const float*)d_in[2];
    const float* wv  = (const float*)d_in[3];
    const float* wo  = (const float*)d_in[4];
    const float* wg  = (const float*)d_in[5];
    const float* cqk = (const float*)d_in[6];
    const float* cqb = (const float*)d_in[7];
    const float* ckk = (const float*)d_in[8];
    const float* ckb = (const float*)d_in[9];
    const float* lrw = (const float*)d_in[10];
    const float* lrbv= (const float*)d_in[11];
    const float* lti = (const float*)d_in[12];
    const float* gam = (const float*)d_in[13];
    const float* bet = (const float*)d_in[14];
    const float* ps  = (const float*)d_in[15];
    const float* pb  = (const float*)d_in[16];
    const float* W1  = (const float*)d_in[17];
    const float* b1  = (const float*)d_in[18];

    float* ws = (float*)d_ws;
    const size_t S  = (size_t)NTOK * H_DIM;       // 6.29M
    const size_t SW = (size_t)H_DIM * H_DIM;      // 589824
    float* xqk = ws;                              // dead after convrope
    float* XVg = ws + 1 * S;
    float* Gg  = ws + 2 * S;
    float* XQs = ws + 3 * S;
    float* XKs = ws + 4 * S;
    float* XVs = ws + 5 * S;
    float* Yb  = ws + 6 * S;
    float* lr  = ws + 7 * S;                      // 98304 floats
    bf16* hsb  = (bf16*)(ws + 7 * S + 98304);     // S bf16
    bf16* wqT  = (bf16*)(ws + 7 * S + 98304 + S / 2);
    bf16* wvT  = wqT + SW;
    bf16* wgT  = wvT + SW;
    bf16* woT  = wgT + SW;
    bf16* tmp  = (bf16*)xqk;                      // reuse

    // one-time bf16 conversions / weight transposes
    cvt_bf16_kernel<<<NTOK * H_DIM / (256 * 8), 256, 0, stream>>>(hs, hsb);
    wT_kernel<<<SW / 256, 256, 0, stream>>>(wq, wqT);
    wT_kernel<<<SW / 256, 256, 0, stream>>>(wv, wvT);
    wT_kernel<<<SW / 256, 256, 0, stream>>>(wg, wgT);
    wT_kernel<<<SW / 256, 256, 0, stream>>>(wo, woT);

    dim3 gg(NTOK / 128, H_DIM / 128, 1);          // 64 x 6
    gemm768_kernel<<<gg, 256, 0, stream>>>(hsb, wqT, xqk);
    gemm768_kernel<<<gg, 256, 0, stream>>>(hsb, wvT, XVg);
    gemm768_kernel<<<gg, 256, 0, stream>>>(hsb, wgT, Gg);

    convrope_kernel<<<dim3(NTOK, 3), 128, 0, stream>>>(
        xqk, XVg, pos, cqk, cqb, ckk, ckb, XQs, XKs, XVs);

    lr_kernel<<<(BATCH * NH * NMB * MB) / 256, 256, 0, stream>>>(hs, lrw, lrbv, lr);

    ttt_scan_kernel<<<BATCH * NH, 32, 0, stream>>>(
        XQs, XKs, XVs, lr, lti, gam, bet, W1, b1, Yb);

    postgate_kernel<<<NTOK, 256, 0, stream>>>(Yb, Gg, ps, pb, tmp);

    gemm768_kernel<<<gg, 256, 0, stream>>>(tmp, woT, (float*)d_out);
}
